// AttentionalCopula_936302870893
// MI455X (gfx1250) — matrix operand order repaired and checked by reading, not compiled, not run on hardware
//
#include <hip/hip_runtime.h>

// ---------------------------------------------------------------------------
// AttentionalCopula forward for MI455X (gfx1250): all GEMMs via
// v_wmma_f32_16x16x32_bf16 (bf16 operands, f32 accumulate).
// All B operands are pre-transposed to [N,K] so both A and B tiles are
// K-contiguous and staged with global_load_async_to_lds_b128 (ASYNCcnt),
// double-buffered in LDS (single barrier per K-step) so WMMA overlaps the
// next tile's global traffic. Each wave computes a 32x32 output block
// (4 WMMA per K-step, 8 ds_load_b128). Fragment feeds are aligned b128.
// ---------------------------------------------------------------------------

typedef __attribute__((ext_vector_type(16))) __bf16 v16bf;
typedef __attribute__((ext_vector_type(8)))  __bf16 v8bf;
typedef __attribute__((ext_vector_type(8)))  float  v8f;

constexpr int Bc  = 16;
constexpr int Dc  = 256;
constexpr int NHc = 512;
constexpr int NVc = 256;
constexpr int Lc  = 4;
constexpr int Hc  = 8;
constexpr int Ac  = 64;
constexpr int HAc = 512;
constexpr int Mc  = 512;
constexpr int Rc  = 128;
constexpr int Wc  = NHc + NVc;   // 768
constexpr int KPc = 288;         // (D+1)=257 padded up to 9*32
constexpr float EPSc = 1e-5f;

static __device__ __forceinline__ float wave_sum(float v) {
#pragma unroll
  for (int m = 16; m >= 1; m >>= 1) v += __shfl_xor(v, m, 32);
  return v;
}
static __device__ __forceinline__ float wave_max(float v) {
#pragma unroll
  for (int m = 16; m >= 1; m >>= 1) v = fmaxf(v, __shfl_xor(v, m, 32));
  return v;
}
static __device__ __forceinline__ v16bf cat16(v8bf lo, v8bf hi) {
  return __builtin_shufflevector(lo, hi, 0, 1, 2, 3, 4, 5, 6, 7, 8, 9, 10, 11,
                                 12, 13, 14, 15);
}

// ---------------------------------------------------------------------------
// Conversion kernels
// ---------------------------------------------------------------------------

// dst[l][n][k] = (bf16) src[l][k][n]   (weights stored transposed, [N,K])
__global__ void cvt_t_kernel(const float* __restrict__ src,
                             __bf16* __restrict__ dst, int Ld, int Kd,
                             int Nd) {
  long long n_total = (long long)Ld * Kd * Nd;
  long long i = (long long)blockIdx.x * blockDim.x + threadIdx.x;
  long long stride = (long long)gridDim.x * blockDim.x;
  long long plane = (long long)Kd * Nd;
  for (; i < n_total; i += stride) {
    long long l = i / plane;
    long long rem = i % plane;
    int n = (int)(rem / Kd);
    int k = (int)(rem % Kd);
    dst[i] = (__bf16)src[(l * Kd + k) * Nd + n];
  }
}

// plain f32 -> bf16 (row-major kept; used for A-side activations)
__global__ void cvt_bf16_kernel(const float* __restrict__ src,
                                __bf16* __restrict__ dst, long long n) {
  long long i = (long long)blockIdx.x * blockDim.x + threadIdx.x;
  long long stride = (long long)gridDim.x * blockDim.x;
  for (; i < n; i += stride) dst[i] = (__bf16)src[i];
}

// key_w/val_w: [L,H,257,A] f32 -> [L*H][A][KPc] bf16 (transposed, K padded)
__global__ void cvt_kvw_t_kernel(const float* __restrict__ src,
                                 __bf16* __restrict__ dst) {
  long long n = (long long)Lc * Hc * Ac * KPc;
  long long i = (long long)blockIdx.x * blockDim.x + threadIdx.x;
  long long stride = (long long)gridDim.x * blockDim.x;
  for (; i < n; i += stride) {
    int k = (int)(i % KPc);
    long long t = i / KPc;
    int a = (int)(t % Ac);
    long long lh = t / Ac;
    dst[i] = (k < Dc + 1) ? (__bf16)src[(lh * (Dc + 1) + k) * Ac + a]
                          : (__bf16)0.0f;
  }
}

// ki = [hist||u ; pred||u] padded: [B, W, KPc] bf16
__global__ void build_ki_kernel(const float* __restrict__ hist,
                                const float* __restrict__ histu,
                                const float* __restrict__ pred,
                                const float* __restrict__ predu,
                                __bf16* __restrict__ ki) {
  long long n = (long long)Bc * Wc * KPc;
  long long i = (long long)blockIdx.x * blockDim.x + threadIdx.x;
  long long stride = (long long)gridDim.x * blockDim.x;
  for (; i < n; i += stride) {
    int k = (int)(i % KPc);
    long long t = i / KPc;
    int w = (int)(t % Wc);
    int b = (int)(t / Wc);
    float v = 0.0f;
    if (k <= Dc) {
      if (w < NHc) {
        v = (k < Dc) ? hist[((long long)b * NHc + w) * Dc + k]
                     : histu[b * NHc + w];
      } else {
        int p = w - NHc;
        v = (k < Dc) ? pred[((long long)b * NVc + p) * Dc + k]
                     : predu[b * NVc + p];
      }
    }
    ki[i] = (__bf16)v;
  }
}

// ---------------------------------------------------------------------------
// Batched bf16 GEMM: C[M,N] = alpha * A[M,K] @ Bt[N,K]^T (+bias) (+relu)
// Block tile 128x64, BK=32, 256 threads = 8 waves; each wave owns a 32x32
// output block: 2 A-frags x 2 B-frags = 4 v_wmma_f32_16x16x32_bf16 per step.
// Double-buffered async staging, one barrier per K-step.
// ctrans!=0 stores C transposed (off = col*ldc + row).
// blockIdx.z = b*Hdim + h with independent b/h strides for A, B, C, bias.
// ---------------------------------------------------------------------------
__global__ __launch_bounds__(256) void gemm_bf16_kernel(
    const __bf16* __restrict__ A, const __bf16* __restrict__ Bt,
    const float* __restrict__ bias, float* __restrict__ Cf,
    __bf16* __restrict__ Cb, int Kdim, int lda, int ldb, int ldc,
    long long sAb, long long sAh, long long sBb, long long sBh, long long sCb,
    long long sCh, int sBiasH, int Hdim, int ctrans, int relu, float alpha) {
  __shared__ __bf16 As[2][128][40];  // [buf][m][k], 80B row pitch
  __shared__ __bf16 Bs[2][64][40];   // [buf][n][k], 80B row pitch

  int z = blockIdx.z;
  int bb = z / Hdim;
  int hh = z % Hdim;
  A  += bb * sAb + hh * sAh;
  Bt += bb * sBb + hh * sBh;
  long long coff = bb * sCb + hh * sCh;
  if (bias) bias += hh * sBiasH;

  int tid = threadIdx.x;
  int lane = tid & 31;
  int wid = tid >> 5;
  int wm = wid & 3;        // wave 32-row group
  int wn = wid >> 2;       // wave 32-col group
  int tileM = blockIdx.y * 128;
  int tileN = blockIdx.x * 64;

  v8f c00 = {0.f, 0.f, 0.f, 0.f, 0.f, 0.f, 0.f, 0.f};
  v8f c01 = c00, c10 = c00, c11 = c00;

  // Staging: A tile 128x32 (2 chunks/thread: rows t>>2 and t>>2+64),
  //          B tile 64x32 (1 chunk/thread). 16B per async op.
  int trow = tid >> 2, tkg = (tid & 3) * 8;
  const __bf16* aBase0 = A + (long long)(tileM + trow) * lda + tkg;
  const __bf16* aBase1 = aBase0 + 64LL * lda;
  const __bf16* bBase  = Bt + (long long)(tileN + trow) * ldb + tkg;
  // LDS byte addresses (generic addr low 32 bits = LDS offset)
  unsigned lA0[2] = {(unsigned)(unsigned long long)&As[0][trow][tkg],
                     (unsigned)(unsigned long long)&As[1][trow][tkg]};
  unsigned lA1[2] = {(unsigned)(unsigned long long)&As[0][trow + 64][tkg],
                     (unsigned)(unsigned long long)&As[1][trow + 64][tkg]};
  unsigned lB[2]  = {(unsigned)(unsigned long long)&Bs[0][trow][tkg],
                     (unsigned)(unsigned long long)&Bs[1][trow][tkg]};

  // Prologue: stream first K-chunk into buffer 0.
  asm volatile(
      "global_load_async_to_lds_b128 %0, %1, off\n\t"
      "global_load_async_to_lds_b128 %2, %3, off\n\t"
      "global_load_async_to_lds_b128 %4, %5, off" ::"v"(lA0[0]),
      "v"(aBase0), "v"(lA1[0]), "v"(aBase1), "v"(lB[0]), "v"(bBase)
      : "memory");

  int nk = Kdim >> 5;
  for (int i = 0; i < nk; ++i) {
    asm volatile("s_wait_asynccnt 0" ::: "memory");
    // Single barrier per step: makes buffer (i&1) visible to all waves AND
    // orders every wave's reads of buffer ((i+1)&1) (done at step i-1)
    // before the writes issued below.
    __syncthreads();
    if (i + 1 < nk) {
      int koff = (i + 1) * 32;
      int nbuf = (i + 1) & 1;
      const __bf16* ap0 = aBase0 + koff;
      const __bf16* ap1 = aBase1 + koff;
      const __bf16* bp  = bBase + koff;
      asm volatile(
          "global_load_async_to_lds_b128 %0, %1, off\n\t"
          "global_load_async_to_lds_b128 %2, %3, off\n\t"
          "global_load_async_to_lds_b128 %4, %5, off" ::"v"(lA0[nbuf]),
          "v"(ap0), "v"(lA1[nbuf]), "v"(ap1), "v"(lB[nbuf]), "v"(bp)
          : "memory");
    }
    int buf = i & 1;

    // A fragments (16x32 ISA layout): lanes 0-15 -> K 0-7 & 16-23,
    // lanes 16-31 -> K 8-15 & 24-31; aligned b128 LDS loads.
    int mrow = wm * 32 + (lane & 15);
    int kbA = (lane < 16) ? 0 : 8;
    v16bf af0 = cat16(*reinterpret_cast<const v8bf*>(&As[buf][mrow][kbA]),
                      *reinterpret_cast<const v8bf*>(&As[buf][mrow][kbA + 16]));
    v16bf af1 =
        cat16(*reinterpret_cast<const v8bf*>(&As[buf][mrow + 16][kbA]),
              *reinterpret_cast<const v8bf*>(&As[buf][mrow + 16][kbA + 16]));
    // B fragments (32x16): lane = column, lanes 0-15 K 0-7 & 16-23, 16-31 K 8-15 & 24-31.
    int n0 = wn * 32 + (lane & 15);
    int kbB = (lane < 16) ? 0 : 8;
    v16bf bf0 = cat16(*reinterpret_cast<const v8bf*>(&Bs[buf][n0][kbB]),
                      *reinterpret_cast<const v8bf*>(&Bs[buf][n0][kbB + 16]));
    v16bf bf1 =
        cat16(*reinterpret_cast<const v8bf*>(&Bs[buf][n0 + 16][kbB]),
              *reinterpret_cast<const v8bf*>(&Bs[buf][n0 + 16][kbB + 16]));

    c00 = __builtin_amdgcn_wmma_f32_16x16x32_bf16(false, af0, false, bf0,
                                                  (short)0, c00, false, false);
    c01 = __builtin_amdgcn_wmma_f32_16x16x32_bf16(false, af0, false, bf1,
                                                  (short)0, c01, false, false);
    c10 = __builtin_amdgcn_wmma_f32_16x16x32_bf16(false, af1, false, bf0,
                                                  (short)0, c10, false, false);
    c11 = __builtin_amdgcn_wmma_f32_16x16x32_bf16(false, af1, false, bf1,
                                                  (short)0, c11, false, false);
  }

  // C/D layout: VGPR r -> row r (lanes 0-15) / r+8 (lanes 16-31), col = lane&15
  int colBase = tileN + wn * 32 + (lane & 15);
  int rowBase = tileM + wm * 32 + ((lane < 16) ? 0 : 8);
#pragma unroll
  for (int r = 0; r < 8; ++r) {
#pragma unroll
    for (int ti = 0; ti < 2; ++ti) {
      int row = rowBase + ti * 16 + r;
#pragma unroll
      for (int tj = 0; tj < 2; ++tj) {
        int col = colBase + tj * 16;
        float v;
        if (ti == 0)
          v = (tj == 0 ? c00[r] : c01[r]);
        else
          v = (tj == 0 ? c10[r] : c11[r]);
        v *= alpha;
        if (bias) v += bias[col];
        if (relu) v = fmaxf(v, 0.f);
        long long off = ctrans ? coff + (long long)col * ldc + row
                               : coff + (long long)row * ldc + col;
        if (Cf) Cf[off] = v;
        if (Cb) Cb[off] = (__bf16)v;
      }
    }
  }
}

// ---------------------------------------------------------------------------
// Masked softmax over rows of W=768: rows = B*H*NV, mask w >= NH + v
// (scores already scaled in the GEMM). Output P in bf16.
// ---------------------------------------------------------------------------
__global__ __launch_bounds__(256) void softmax_mask_kernel(
    const float* __restrict__ S, __bf16* __restrict__ P) {
  int lane = threadIdx.x & 31;
  int wid = threadIdx.x >> 5;
  long long row = (long long)blockIdx.x * 8 + wid;
  if (row >= (long long)Bc * Hc * NVc) return;
  int v = (int)(row % NVc);
  const float* s = S + row * Wc;
  float vals[Wc / 32];
  float mx = -3.0e38f;
#pragma unroll
  for (int j = 0; j < Wc / 32; ++j) {
    int w = lane + j * 32;
    float x = s[w];
    if (w >= NHc + v) x = -__builtin_inff();
    vals[j] = x;
    mx = fmaxf(mx, x);
  }
  mx = wave_max(mx);
  float sum = 0.f;
#pragma unroll
  for (int j = 0; j < Wc / 32; ++j) {
    float e = __expf(vals[j] - mx);  // exp(-inf) -> 0 for masked
    vals[j] = e;
    sum += e;
  }
  sum = wave_sum(sum);
  float inv = 1.0f / sum;
  __bf16* p = P + row * Wc;
#pragma unroll
  for (int j = 0; j < Wc / 32; ++j) p[lane + j * 32] = (__bf16)(vals[j] * inv);
}

// ---------------------------------------------------------------------------
// LayerNorm(residual): y = g * norm(x + r) + b ; rows = B*NV of HA=512.
// One wave per row; dual f32 + bf16 store.
// ---------------------------------------------------------------------------
__global__ __launch_bounds__(256) void ln_residual_kernel(
    const float* __restrict__ x, const float* __restrict__ r,
    const float* __restrict__ g, const float* __restrict__ bta,
    float* __restrict__ y, __bf16* __restrict__ yb) {
  int lane = threadIdx.x & 31;
  int wid = threadIdx.x >> 5;
  long long row = (long long)blockIdx.x * 8 + wid;
  if (row >= (long long)Bc * NVc) return;
  const float* xr = x + row * HAc;
  const float* rr = r + row * HAc;
  float v[HAc / 32];
  float s = 0.f;
#pragma unroll
  for (int j = 0; j < HAc / 32; ++j) {
    v[j] = xr[lane + j * 32] + rr[lane + j * 32];
    s += v[j];
  }
  float mean = wave_sum(s) * (1.0f / HAc);
  float q = 0.f;
#pragma unroll
  for (int j = 0; j < HAc / 32; ++j) {
    float d = v[j] - mean;
    q += d * d;
  }
  float var = wave_sum(q) * (1.0f / HAc);
  float inv = rsqrtf(var + EPSc);
#pragma unroll
  for (int j = 0; j < HAc / 32; ++j) {
    int c = lane + j * 32;
    float o = g[c] * (v[j] - mean) * inv + bta[c];
    y[row * HAc + c] = o;
    yb[row * HAc + c] = (__bf16)o;
  }
}

// ---------------------------------------------------------------------------
// Final NLL: out[b] = -sum_{v=1..NV-1} (log R + logits[b,v,tgt] - lse(row))
// ---------------------------------------------------------------------------
__global__ __launch_bounds__(256) void loss_kernel(
    const float* __restrict__ logits, const float* __restrict__ predu,
    float* __restrict__ out) {
  int b = blockIdx.x;
  int lane = threadIdx.x & 31;
  int wid = threadIdx.x >> 5;
  __shared__ float part[8];
  float acc = 0.f;
  const float logR = __logf((float)Rc);
  for (int v = 1 + wid; v < NVc; v += 8) {
    const float* lr = logits + ((long long)b * NVc + v) * Rc;
    float x[Rc / 32];
    float mx = -3.0e38f;
#pragma unroll
    for (int j = 0; j < Rc / 32; ++j) {
      x[j] = lr[lane + j * 32];
      mx = fmaxf(mx, x[j]);
    }
    mx = wave_max(mx);
    float s = 0.f;
#pragma unroll
    for (int j = 0; j < Rc / 32; ++j) s += __expf(x[j] - mx);
    s = wave_sum(s);
    float lse = mx + __logf(s);
    int tgt = (int)floorf(predu[b * NVc + v] * (float)Rc);
    tgt = min(max(tgt, 0), Rc - 1);
    float lt = lr[tgt];
    if (lane == 0) acc -= (logR + lt - lse);
  }
  if (lane == 0) part[wid] = acc;
  __syncthreads();
  if (threadIdx.x == 0) {
    float t = 0.f;
#pragma unroll
    for (int i = 0; i < 8; ++i) t += part[i];
    out[b] = t;
  }
}

// ---------------------------------------------------------------------------
// Host launcher
// ---------------------------------------------------------------------------
extern "C" void kernel_launch(void* const* d_in, const int* in_sizes, int n_in,
                              void* d_out, int out_size, void* d_ws,
                              size_t ws_size, hipStream_t stream) {
  (void)in_sizes; (void)n_in; (void)out_size; (void)ws_size;
  const float* hist  = (const float*)d_in[0];
  const float* histu = (const float*)d_in[1];
  const float* pred  = (const float*)d_in[2];
  const float* predu = (const float*)d_in[3];
  const float* ds_w  = (const float*)d_in[4];
  const float* ds_b  = (const float*)d_in[5];
  const float* key_w = (const float*)d_in[6];
  const float* key_b = (const float*)d_in[7];
  const float* val_w = (const float*)d_in[8];
  const float* val_b = (const float*)d_in[9];
  const float* ln1_g = (const float*)d_in[10];
  const float* ln1_b = (const float*)d_in[11];
  const float* ln2_g = (const float*)d_in[12];
  const float* ln2_b = (const float*)d_in[13];
  const float* ff_w1 = (const float*)d_in[14];
  const float* ff_b1 = (const float*)d_in[15];
  const float* ff_w2 = (const float*)d_in[16];
  const float* ff_b2 = (const float*)d_in[17];
  const float* ff_w3 = (const float*)d_in[18];
  const float* ff_b3 = (const float*)d_in[19];
  const float* de_w  = (const float*)d_in[20];
  const float* de_b  = (const float*)d_in[21];
  float* out = (float*)d_out;

  // ---- workspace carve ----
  char* ws = (char*)d_ws;
  size_t off = 0;
  auto carve = [&](size_t bytes) -> char* {
    char* p = ws + off;
    off += (bytes + 255) & ~(size_t)255;
    return p;
  };
  __bf16* pred_bf = (__bf16*)carve((size_t)Bc * NVc * Dc * 2);
  __bf16* ki_bf   = (__bf16*)carve((size_t)Bc * Wc * KPc * 2);
  __bf16* dsw_t   = (__bf16*)carve((size_t)HAc * Dc * 2);        // [HA][D]
  __bf16* keyw_t  = (__bf16*)carve((size_t)Lc * Hc * Ac * KPc * 2);  // [l,h][A][KP]
  __bf16* valw_t  = (__bf16*)carve((size_t)Lc * Hc * Ac * KPc * 2);
  __bf16* ffw1_t  = (__bf16*)carve((size_t)Lc * Mc * HAc * 2);   // [l][M][HA]
  __bf16* ffw2_t  = (__bf16*)carve((size_t)Lc * Mc * Mc * 2);    // [l][M][M]
  __bf16* ffw3_t  = (__bf16*)carve((size_t)Lc * HAc * Mc * 2);   // [l][HA][M]
  __bf16* dew_t   = (__bf16*)carve((size_t)Rc * HAc * 2);        // [R][HA]
  float*  att_f   = (float*)carve((size_t)Bc * NVc * HAc * 4);
  __bf16* att_b   = (__bf16*)carve((size_t)Bc * NVc * HAc * 2);
  __bf16* keys_b  = (__bf16*)carve((size_t)Hc * Bc * Wc * Ac * 2);  // [H,B,W,A]
  __bf16* vals_t  = (__bf16*)carve((size_t)Hc * Bc * Ac * Wc * 2);  // [H,B,A,W]
  float*  scores  = (float*)carve((size_t)Bc * Hc * NVc * Wc * 4);
  __bf16* P_b     = (__bf16*)carve((size_t)Bc * Hc * NVc * Wc * 2);
  float*  atta_f  = (float*)carve((size_t)Bc * NVc * HAc * 4);
  __bf16* ff1_b   = (__bf16*)carve((size_t)Bc * NVc * Mc * 2);
  __bf16* ff2_b   = (__bf16*)carve((size_t)Bc * NVc * Mc * 2);
  float*  ff3_f   = (float*)carve((size_t)Bc * NVc * HAc * 4);
  float*  logits  = (float*)carve((size_t)Bc * NVc * Rc * 4);

  auto gemm = [&](const __bf16* A, const __bf16* Bt, const float* bias,
                  float* Cf, __bf16* Cb, int M, int N, int K, int lda, int ldb,
                  int ldc, long long sAb, long long sAh, long long sBb,
                  long long sBh, long long sCb, long long sCh, int sBiasH,
                  int Hd, int nz, int ctrans, int relu, float alpha) {
    dim3 g(N / 64, M / 128, nz), blk(256);
    gemm_bf16_kernel<<<g, blk, 0, stream>>>(A, Bt, bias, Cf, Cb, K, lda, ldb,
                                            ldc, sAb, sAh, sBb, sBh, sCb, sCh,
                                            sBiasH, Hd, ctrans, relu, alpha);
  };

  // ---- prologue: conversions (weights transposed to [N,K]) ----
  cvt_bf16_kernel<<<2048, 256, 0, stream>>>(pred, pred_bf,
                                            (long long)Bc * NVc * Dc);
  build_ki_kernel<<<2048, 256, 0, stream>>>(hist, histu, pred, predu, ki_bf);
  cvt_t_kernel<<<512, 256, 0, stream>>>(ds_w, dsw_t, 1, Dc, HAc);
  cvt_kvw_t_kernel<<<1024, 256, 0, stream>>>(key_w, keyw_t);
  cvt_kvw_t_kernel<<<1024, 256, 0, stream>>>(val_w, valw_t);
  cvt_t_kernel<<<2048, 256, 0, stream>>>(ff_w1, ffw1_t, Lc, HAc, Mc);
  cvt_t_kernel<<<2048, 256, 0, stream>>>(ff_w2, ffw2_t, Lc, Mc, Mc);
  cvt_t_kernel<<<2048, 256, 0, stream>>>(ff_w3, ffw3_t, Lc, Mc, HAc);
  cvt_t_kernel<<<256, 256, 0, stream>>>(de_w, dew_t, 1, HAc, Rc);

  // ---- dimension-shift: att = pred @ ds_w + ds_b ----
  gemm(pred_bf, dsw_t, ds_b, att_f, att_b, Bc * NVc, HAc, Dc, Dc, Dc, HAc, 0,
       0, 0, 0, 0, 0, 0, 1, 1, 0, 0, 1.0f);

  const float scale = 0.125f;  // A^-0.5 = 1/8
  for (int l = 0; l < Lc; ++l) {
    // keys: per (b,h): ki[b] [W,KP] @ key_w[l,h]^T -> keys_b [H,B,W,A]
    gemm(ki_bf, keyw_t + (size_t)l * Hc * Ac * KPc, key_b + l * Hc * Ac,
         nullptr, keys_b, Wc, Ac, KPc, KPc, KPc, Ac, (long long)Wc * KPc, 0, 0,
         (long long)Ac * KPc, (long long)Wc * Ac, (long long)Bc * Wc * Ac, Ac,
         Hc, Bc * Hc, 0, 0, 1.0f);
    // vals: same, but stored transposed -> vals_t [H,B,A,W]
    gemm(ki_bf, valw_t + (size_t)l * Hc * Ac * KPc, val_b + l * Hc * Ac,
         nullptr, vals_t, Wc, Ac, KPc, KPc, KPc, Wc, (long long)Wc * KPc, 0, 0,
         (long long)Ac * KPc, (long long)Ac * Wc, (long long)Bc * Ac * Wc, Ac,
         Hc, Bc * Hc, 1, 0, 1.0f);
    // scores = scale * Q @ K^T  (Q = att slice, Bt = keys[h][b] rows=w,K=a)
    gemm(att_b, keys_b, nullptr, scores, nullptr, NVc, Wc, Ac, HAc, Ac, Wc,
         (long long)NVc * HAc, Ac, (long long)Wc * Ac, (long long)Bc * Wc * Ac,
         (long long)Hc * NVc * Wc, (long long)NVc * Wc, 0, Hc, Bc * Hc, 0, 0,
         scale);
    // masked softmax -> P (bf16)
    softmax_mask_kernel<<<(Bc * Hc * NVc) / 8, 256, 0, stream>>>(scores, P_b);
    // a = P @ V  (Bt = vals_t[h][b] rows=a,K=w) -> atta_f[b, v, h*64 + a]
    gemm(P_b, vals_t, nullptr, atta_f, nullptr, NVc, Ac, Wc, Wc, Wc, HAc,
         (long long)Hc * NVc * Wc, (long long)NVc * Wc, (long long)Ac * Wc,
         (long long)Bc * Ac * Wc, (long long)NVc * HAc, Ac, 0, Hc, Bc * Hc, 0,
         0, 1.0f);
    // att = LN1(att + a)
    ln_residual_kernel<<<(Bc * NVc) / 8, 256, 0, stream>>>(
        att_f, atta_f, ln1_g + l * HAc, ln1_b + l * HAc, att_f, att_b);
    // FFN
    gemm(att_b, ffw1_t + (size_t)l * Mc * HAc, ff_b1 + l * Mc, nullptr, ff1_b,
         Bc * NVc, Mc, HAc, HAc, HAc, Mc, 0, 0, 0, 0, 0, 0, 0, 1, 1, 0, 1,
         1.0f);
    gemm(ff1_b, ffw2_t + (size_t)l * Mc * Mc, ff_b2 + l * Mc, nullptr, ff2_b,
         Bc * NVc, Mc, Mc, Mc, Mc, Mc, 0, 0, 0, 0, 0, 0, 0, 1, 1, 0, 1, 1.0f);
    gemm(ff2_b, ffw3_t + (size_t)l * HAc * Mc, ff_b3 + l * HAc, ff3_f, nullptr,
         Bc * NVc, HAc, Mc, Mc, Mc, HAc, 0, 0, 0, 0, 0, 0, 0, 1, 1, 0, 0,
         1.0f);
    // att = LN2(att + ff)
    ln_residual_kernel<<<(Bc * NVc) / 8, 256, 0, stream>>>(
        att_f, ff3_f, ln2_g + l * HAc, ln2_b + l * HAc, att_f, att_b);
  }

  // ---- logits + loss ----
  gemm(att_b, dew_t, de_b, logits, nullptr, Bc * NVc, Rc, HAc, HAc, HAc, Rc, 0,
       0, 0, 0, 0, 0, 0, 1, 1, 0, 0, 1.0f);
  loss_kernel<<<Bc, 256, 0, stream>>>(logits, predu, out);
}
